// CausalSelfAttention_54924041781785
// MI455X (gfx1250) — matrix operand order repaired and checked by reading, not compiled, not run on hardware
//
#include <hip/hip_runtime.h>

// ---------------------------------------------------------------------------
// CDNA5 (gfx1250) wave32 WMMA attention block.
// B=4, S=2048, C=768, H=12, D=64.  bf16 WMMA math, f32 accumulate.
// Double-buffered LDS pipelines; raw bf16 tiles use global_load_async_to_lds.
// ---------------------------------------------------------------------------

typedef __attribute__((ext_vector_type(16))) __bf16 bf16x16;
typedef __attribute__((ext_vector_type(8)))  float  f32x8;

union U16x16 { uint4 q[2]; bf16x16 v; };
union U8x16  { uint4 q;    __bf16 e[8]; };

static __device__ __forceinline__ f32x8 wmma_bf16(bf16x16 a, bf16x16 b, f32x8 c) {
  return __builtin_amdgcn_wmma_f32_16x16x32_bf16(false, a, false, b, (short)0, c,
                                                 false, false);
}

// Async 16B copy global -> LDS (ASYNCcnt tracked, no VGPR data movement).
static __device__ __forceinline__ void async_copy_b128(void* lds_dst,
                                                       const void* gsrc) {
  const unsigned l = (unsigned)(unsigned long long)lds_dst;  // LDS offset = low 32b
  asm volatile("global_load_async_to_lds_b128 %0, %1, off"
               :: "v"(l), "v"(gsrc)
               : "memory");
}
static __device__ __forceinline__ void wait_async0() {
  asm volatile("s_wait_asynccnt 0" ::: "memory");
}

// A fragment, 16x32 bf16, from row-major LDS (row = lane&15): 2x ds_load_b128.
static __device__ __forceinline__ bf16x16
lds_load_a16x32(const __bf16* base, int stride, int lane) {
  const int row = lane & 15;
  const int kb  = (lane >> 4) << 3;
  const __bf16* p = base + row * stride + kb;
  U16x16 u;
  u.q[0] = *reinterpret_cast<const uint4*>(p);        // K kb .. kb+7
  u.q[1] = *reinterpret_cast<const uint4*>(p + 16);   // K kb+16 .. kb+23
  return u.v;
}

// B fragment from an N-major (pre-transposed) LDS tile: 2x ds_load_b128.
static __device__ __forceinline__ bf16x16
lds_load_bT16(const __bf16* base, int stride, int lane) {
  const int n  = lane & 15;
  const int kb = (lane >> 4) << 3;
  const __bf16* p = base + n * stride + kb;
  U16x16 u;
  u.q[0] = *reinterpret_cast<const uint4*>(p);
  u.q[1] = *reinterpret_cast<const uint4*>(p + 16);
  return u.v;
}

static __device__ __forceinline__ void cvt8_store(__bf16* dst, float4 a0, float4 a1) {
  U8x16 u;
  u.e[0] = (__bf16)a0.x; u.e[1] = (__bf16)a0.y;
  u.e[2] = (__bf16)a0.z; u.e[3] = (__bf16)a0.w;
  u.e[4] = (__bf16)a1.x; u.e[5] = (__bf16)a1.y;
  u.e[6] = (__bf16)a1.z; u.e[7] = (__bf16)a1.w;
  *reinterpret_cast<uint4*>(dst) = u.q;
}

// ---------------------------------------------------------------------------
// Kernel 1: qkv = x @ Wqkv, fused RoPE, scatter to (B,H,S,D) bf16.
// Grid (36, 64), block 256.  128x64 tile; wave = 32x32.  Double-buffered LDS,
// register prefetch (both operands need f32->bf16 conversion).
// ---------------------------------------------------------------------------
__global__ __launch_bounds__(256) void qkv_rope_kernel(
    const float* __restrict__ x, const float* __restrict__ Wqkv,
    __bf16* __restrict__ qo, __bf16* __restrict__ ko, __bf16* __restrict__ vo) {
  __shared__ __bf16 lx[2][128 * 40];   // row-major x tiles
  __shared__ __bf16 lwT[2][64 * 40];   // N-major W tiles
  __shared__ __bf16 lo[128 * 72];      // RoPE staging

  const int t    = threadIdx.x;
  const int lane = t & 31;
  const int wid  = t >> 5;
  const int wrow = wid >> 1;
  const int wcol = wid & 1;
  const int hh   = lane >> 4;
  const int ln   = lane & 15;
  const int bx   = blockIdx.x;
  const int by   = blockIdx.y;

  const int xr_ = t >> 1, xc = (t & 1) << 4;     // x-tile fill coords
  const int wk  = t >> 3, wn = (t & 7) << 3;     // W-tile fill coords
  const float* xg = x + (size_t)(by * 128 + xr_) * 768 + xc;
  const float* wg = Wqkv + (size_t)wk * 2304 + bx * 64 + wn;

  float4 xv[4], wv[2];
  auto g2r = [&](int k0) {
    const float4* gx = reinterpret_cast<const float4*>(xg + k0);
    xv[0] = gx[0]; xv[1] = gx[1]; xv[2] = gx[2]; xv[3] = gx[3];
    const float4* gw = reinterpret_cast<const float4*>(wg + (size_t)k0 * 2304);
    wv[0] = gw[0]; wv[1] = gw[1];
  };
  auto r2s = [&](int buf) {
    cvt8_store(&lx[buf][xr_ * 40 + xc], xv[0], xv[1]);
    cvt8_store(&lx[buf][xr_ * 40 + xc + 8], xv[2], xv[3]);
    __bf16* dw = &lwT[buf][0];
    dw[(wn + 0) * 40 + wk] = (__bf16)wv[0].x;
    dw[(wn + 1) * 40 + wk] = (__bf16)wv[0].y;
    dw[(wn + 2) * 40 + wk] = (__bf16)wv[0].z;
    dw[(wn + 3) * 40 + wk] = (__bf16)wv[0].w;
    dw[(wn + 4) * 40 + wk] = (__bf16)wv[1].x;
    dw[(wn + 5) * 40 + wk] = (__bf16)wv[1].y;
    dw[(wn + 6) * 40 + wk] = (__bf16)wv[1].z;
    dw[(wn + 7) * 40 + wk] = (__bf16)wv[1].w;
  };

  f32x8 acc[2][2] = {};

  g2r(0);
  r2s(0);
  __syncthreads();

  for (int ks = 0; ks < 24; ++ks) {
    const int cur = ks & 1;
    if (ks + 1 < 24) g2r((ks + 1) * 32);
    const bf16x16 a0 = lds_load_a16x32(&lx[cur][(wrow * 32) * 40], 40, lane);
    const bf16x16 a1 = lds_load_a16x32(&lx[cur][(wrow * 32 + 16) * 40], 40, lane);
    const bf16x16 b0 = lds_load_bT16(&lwT[cur][(wcol * 32) * 40], 40, lane);
    const bf16x16 b1 = lds_load_bT16(&lwT[cur][(wcol * 32 + 16) * 40], 40, lane);
    acc[0][0] = wmma_bf16(a0, b0, acc[0][0]);
    acc[0][1] = wmma_bf16(a0, b1, acc[0][1]);
    acc[1][0] = wmma_bf16(a1, b0, acc[1][0]);
    acc[1][1] = wmma_bf16(a1, b1, acc[1][1]);
    if (ks + 1 < 24) r2s(cur ^ 1);
    __syncthreads();
  }

  // Stage 128x64 result so RoPE can pair d with d^32.
#pragma unroll
  for (int ar = 0; ar < 2; ++ar)
#pragma unroll
    for (int i = 0; i < 8; ++i) {
      const int row = wrow * 32 + ar * 16 + i + 8 * hh;
      lo[row * 72 + wcol * 32 + ln]      = (__bf16)acc[ar][0][i];
      lo[row * 72 + wcol * 32 + 16 + ln] = (__bf16)acc[ar][1][i];
    }
  __syncthreads();

  // RoPE + scatter.  bx -> (section, head): 768/64 = 12.
  const int sec = bx / 12;
  const int h   = bx % 12;
  const int r   = t >> 1;
  const int cb  = (t & 1) << 5;
  const int rg  = by * 128 + r;
  const int bb  = rg >> 11;
  const int s   = rg & 2047;
  __bf16* dst = ((sec == 0) ? qo : (sec == 1) ? ko : vo) +
                ((size_t)(bb * 12 + h) * 2048 + s) * 64;
#pragma unroll
  for (int j = 0; j < 32; ++j) {
    const int d = cb + j;
    float val = (float)lo[r * 72 + d];
    if (sec < 2) {
      const float partner = (float)lo[r * 72 + (d ^ 32)];
      const float theta = __powf(10000.f, -(float)(d & 31) * (1.f / 32.f));
      float sn, cs;
      __sincosf((float)s * theta, &sn, &cs);
      val = val * cs + ((d < 32) ? -partner : partner) * sn;
    }
    dst[d] = (__bf16)val;
  }
}

// ---------------------------------------------------------------------------
// Kernel 2: causal flash attention.  Grid (16, 48), block 256 (8 waves).
// Double-buffered K/V: K via async-to-LDS, V via register prefetch+transpose.
// ---------------------------------------------------------------------------
__global__ __launch_bounds__(256) void flash_attn_kernel(
    const __bf16* __restrict__ q, const __bf16* __restrict__ k,
    const __bf16* __restrict__ v, __bf16* __restrict__ y) {
  __shared__ __bf16 Qs[128 * 72];       // row-major (s, d)
  __shared__ __bf16 Ks[2][64 * 72];     // row-major (s, d)
  __shared__ __bf16 Vt[2][64 * 72];     // transposed (d, s)
  __shared__ __bf16 Ps[8 * 16 * 72];    // wave-private P strips

  const int t    = threadIdx.x;
  const int lane = t & 31;
  const int w    = t >> 5;
  const int hh   = lane >> 4;
  const int ln   = lane & 15;
  const int qt   = blockIdx.x;
  const int bh   = blockIdx.y;
  const int bb   = bh / 12;
  const int h    = bh % 12;

  const int kr = t >> 2, kd = (t & 3) << 4;  // K/V tile fill coords
  const __bf16* kg = k + ((size_t)bh * 2048 + kr) * 64 + kd;
  const __bf16* vg = v + ((size_t)bh * 2048 + kr) * 64 + kd;

  {  // Q tile 128x64: async copy (raw bf16)
    const int r = t >> 1, db = (t & 1) << 5;
    const __bf16* g = q + ((size_t)bh * 2048 + qt * 128 + r) * 64 + db;
    __bf16* l = &Qs[r * 72 + db];
    async_copy_b128(l, g);
    async_copy_b128(l + 8, g + 8);
    async_copy_b128(l + 16, g + 16);
    async_copy_b128(l + 24, g + 24);
  }

  auto k_async = [&](int kt, int buf) {
    const __bf16* g = kg + (size_t)kt * 64 * 64;
    __bf16* l = &Ks[buf][kr * 72 + kd];
    async_copy_b128(l, g);
    async_copy_b128(l + 8, g + 8);
  };
  U16x16 vpref;
  auto v_g2r = [&](int kt) {
    const uint4* g = reinterpret_cast<const uint4*>(vg + (size_t)kt * 64 * 64);
    vpref.q[0] = g[0];
    vpref.q[1] = g[1];
  };
  auto v_r2s = [&](int buf) {  // transpose scatter Vt[d][s]
#pragma unroll
    for (int j = 0; j < 16; ++j) Vt[buf][(kd + j) * 72 + kr] = vpref.v[j];
  };

  // Preload tile 0.
  k_async(0, 0);
  v_g2r(0);
  v_r2s(0);
  wait_async0();
  __syncthreads();

  const bf16x16 aQ0 = lds_load_a16x32(&Qs[w * 16 * 72], 72, lane);
  const bf16x16 aQ1 = lds_load_a16x32(&Qs[w * 16 * 72 + 32], 72, lane);

  f32x8 oacc[4] = {};
  float m[8], l_[8];
#pragma unroll
  for (int i = 0; i < 8; ++i) { m[i] = -1e30f; l_[i] = 0.f; }

  const float scale = 0.125f;  // 1/sqrt(64)
  const int   qbase = qt * 128 + w * 16 + 8 * hh;
  __bf16* Pw = &Ps[w * 16 * 72];

  const int nkt = 2 * qt + 2;  // causal: key tiles 0 .. 2*qt+1
  for (int kt = 0; kt < nkt; ++kt) {
    const int cur = kt & 1;
    if (kt + 1 < nkt) {  // start next tile's data movement before compute
      k_async(kt + 1, cur ^ 1);
      v_g2r(kt + 1);
    }

    // S = Q K^T  (16 rows x 64 keys per wave; 8 WMMAs)
    f32x8 sacc[4] = {};
#pragma unroll
    for (int nt = 0; nt < 4; ++nt) {
      bf16x16 b0 = lds_load_bT16(&Ks[cur][(nt * 16) * 72], 72, lane);
      sacc[nt] = wmma_bf16(aQ0, b0, sacc[nt]);
      bf16x16 b1 = lds_load_bT16(&Ks[cur][(nt * 16) * 72 + 32], 72, lane);
      sacc[nt] = wmma_bf16(aQ1, b1, sacc[nt]);
    }

    // Online softmax (row reductions across 16 lanes of each half-wave).
    const int kbase = kt * 64 + ln;
#pragma unroll
    for (int i = 0; i < 8; ++i) {
      const int qrow = qbase + i;
      float sv[4];
      float rmax = -1e30f;
#pragma unroll
      for (int nt = 0; nt < 4; ++nt) {
        float val = sacc[nt][i] * scale;
        val = (kbase + nt * 16 <= qrow) ? val : -1e30f;
        sv[nt] = val;
        rmax = fmaxf(rmax, val);
      }
      rmax = fmaxf(rmax, __shfl_xor(rmax, 1, 32));
      rmax = fmaxf(rmax, __shfl_xor(rmax, 2, 32));
      rmax = fmaxf(rmax, __shfl_xor(rmax, 4, 32));
      rmax = fmaxf(rmax, __shfl_xor(rmax, 8, 32));
      const float mnew  = fmaxf(m[i], rmax);
      const float alpha = __expf(m[i] - mnew);
      m[i] = mnew;
      float rsum = 0.f;
#pragma unroll
      for (int nt = 0; nt < 4; ++nt) {
        const float p = __expf(sv[nt] - mnew);
        rsum += p;
        Pw[(i + 8 * hh) * 72 + nt * 16 + ln] = (__bf16)p;
      }
      rsum += __shfl_xor(rsum, 1, 32);
      rsum += __shfl_xor(rsum, 2, 32);
      rsum += __shfl_xor(rsum, 4, 32);
      rsum += __shfl_xor(rsum, 8, 32);
      l_[i] = l_[i] * alpha + rsum;
#pragma unroll
      for (int nt = 0; nt < 4; ++nt) oacc[nt][i] *= alpha;
    }

    // O += P V  (wave-private P strip; LDS ops in-order within a wave).
    const bf16x16 aP0 = lds_load_a16x32(Pw, 72, lane);
    const bf16x16 aP1 = lds_load_a16x32(Pw + 32, 72, lane);
#pragma unroll
    for (int nt = 0; nt < 4; ++nt) {
      bf16x16 bv0 = lds_load_bT16(&Vt[cur][(nt * 16) * 72], 72, lane);
      oacc[nt] = wmma_bf16(aP0, bv0, oacc[nt]);
      bf16x16 bv1 = lds_load_bT16(&Vt[cur][(nt * 16) * 72 + 32], 72, lane);
      oacc[nt] = wmma_bf16(aP1, bv1, oacc[nt]);
    }

    if (kt + 1 < nkt) v_r2s(cur ^ 1);
    wait_async0();
    __syncthreads();
  }

  // Normalize and store y in (B,S,C) bf16 for the projection GEMM.
  __bf16* yb = y + ((size_t)bb * 2048 + qt * 128 + w * 16 + 8 * hh) * 768 + h * 64;
#pragma unroll
  for (int i = 0; i < 8; ++i) {
    const float inv = 1.f / l_[i];
#pragma unroll
    for (int nt = 0; nt < 4; ++nt)
      yb[(size_t)i * 768 + nt * 16 + ln] = (__bf16)(oacc[nt][i] * inv);
  }
}

// ---------------------------------------------------------------------------
// Kernel 3: out = y @ Wproj, f32 output.  Grid (12, 64), block 256.
// Double-buffered: A (bf16) tiles async-to-LDS, W via register prefetch.
// ---------------------------------------------------------------------------
__global__ __launch_bounds__(256) void proj_kernel(
    const __bf16* __restrict__ y, const float* __restrict__ Wp,
    float* __restrict__ out) {
  __shared__ __bf16 la[2][128 * 40];
  __shared__ __bf16 lwT[2][64 * 40];   // N-major

  const int t    = threadIdx.x;
  const int lane = t & 31;
  const int wid  = t >> 5;
  const int wrow = wid >> 1;
  const int wcol = wid & 1;
  const int hh   = lane >> 4;
  const int ln   = lane & 15;
  const int bx   = blockIdx.x;
  const int by   = blockIdx.y;

  const int ar_ = t >> 1, ac = (t & 1) << 4;
  const int wk  = t >> 3, wn = (t & 7) << 3;
  const __bf16* ag = y + (size_t)(by * 128 + ar_) * 768 + ac;
  const float*  wg = Wp + (size_t)wk * 768 + bx * 64 + wn;

  auto a_async = [&](int k0, int buf) {
    const __bf16* g = ag + k0;
    __bf16* l = &la[buf][ar_ * 40 + ac];
    async_copy_b128(l, g);
    async_copy_b128(l + 8, g + 8);
  };
  float4 wv[2];
  auto w_g2r = [&](int k0) {
    const float4* g = reinterpret_cast<const float4*>(wg + (size_t)k0 * 768);
    wv[0] = g[0]; wv[1] = g[1];
  };
  auto w_r2s = [&](int buf) {
    __bf16* dw = &lwT[buf][0];
    dw[(wn + 0) * 40 + wk] = (__bf16)wv[0].x;
    dw[(wn + 1) * 40 + wk] = (__bf16)wv[0].y;
    dw[(wn + 2) * 40 + wk] = (__bf16)wv[0].z;
    dw[(wn + 3) * 40 + wk] = (__bf16)wv[0].w;
    dw[(wn + 4) * 40 + wk] = (__bf16)wv[1].x;
    dw[(wn + 5) * 40 + wk] = (__bf16)wv[1].y;
    dw[(wn + 6) * 40 + wk] = (__bf16)wv[1].z;
    dw[(wn + 7) * 40 + wk] = (__bf16)wv[1].w;
  };

  f32x8 acc[2][2] = {};

  a_async(0, 0);
  w_g2r(0);
  w_r2s(0);
  wait_async0();
  __syncthreads();

  for (int ks = 0; ks < 24; ++ks) {
    const int cur = ks & 1;
    if (ks + 1 < 24) {
      a_async((ks + 1) * 32, cur ^ 1);
      w_g2r((ks + 1) * 32);
    }
    const bf16x16 a0 = lds_load_a16x32(&la[cur][(wrow * 32) * 40], 40, lane);
    const bf16x16 a1 = lds_load_a16x32(&la[cur][(wrow * 32 + 16) * 40], 40, lane);
    const bf16x16 b0 = lds_load_bT16(&lwT[cur][(wcol * 32) * 40], 40, lane);
    const bf16x16 b1 = lds_load_bT16(&lwT[cur][(wcol * 32 + 16) * 40], 40, lane);
    acc[0][0] = wmma_bf16(a0, b0, acc[0][0]);
    acc[0][1] = wmma_bf16(a0, b1, acc[0][1]);
    acc[1][0] = wmma_bf16(a1, b0, acc[1][0]);
    acc[1][1] = wmma_bf16(a1, b1, acc[1][1]);
    if (ks + 1 < 24) w_r2s(cur ^ 1);
    wait_async0();
    __syncthreads();
  }

#pragma unroll
  for (int ar = 0; ar < 2; ++ar) {
    float* ob = out + (size_t)(by * 128 + wrow * 32 + ar * 16 + 8 * hh) * 768 +
                bx * 64 + wcol * 32;
#pragma unroll
    for (int i = 0; i < 8; ++i) {
      ob[(size_t)i * 768 + ln]      = acc[ar][0][i];
      ob[(size_t)i * 768 + 16 + ln] = acc[ar][1][i];
    }
  }
}

// ---------------------------------------------------------------------------
extern "C" void kernel_launch(void* const* d_in, const int* in_sizes, int n_in,
                              void* d_out, int out_size, void* d_ws, size_t ws_size,
                              hipStream_t stream) {
  (void)in_sizes; (void)n_in; (void)out_size; (void)ws_size;
  const float* x     = (const float*)d_in[0];  // (4,2048,768)
  const float* Wqkv  = (const float*)d_in[1];  // (768,2304)
  const float* Wproj = (const float*)d_in[2];  // (768,768)
  float* out = (float*)d_out;                  // (4,2048,768) f32

  const size_t QKV_ELEMS = (size_t)4 * 12 * 2048 * 64;  // 6,291,456
  __bf16* q = (__bf16*)d_ws;
  __bf16* k = q + QKV_ELEMS;
  __bf16* v = k + QKV_ELEMS;
  __bf16* y = v + QKV_ELEMS;  // (B,S,C) bf16

  qkv_rope_kernel<<<dim3(36, 64), 256, 0, stream>>>(x, Wqkv, q, k, v);
  flash_attn_kernel<<<dim3(16, 48), 256, 0, stream>>>(q, k, v, y);
  proj_kernel<<<dim3(12, 64), 256, 0, stream>>>(y, Wproj, out);
}
